// StockHeteGAT_79972291052115
// MI455X (gfx1250) — matrix operand order repaired and checked by reading, not compiled, not run on hardware
//
#include <hip/hip_runtime.h>

typedef __attribute__((ext_vector_type(16))) _Float16 v16h;
typedef __attribute__((ext_vector_type(8)))  _Float16 v8h;
typedef __attribute__((ext_vector_type(8)))  float    v8f;
typedef int b128_t __attribute__((vector_size(16)));   // matches builtin's V4i param

#define N_NODES 3072
#define SEQ     32
#define FIN_    8
#define DIM     128
#define DFF_    512
#define GH      8     // GAT heads
#define GFO     16    // GAT out features per head
#define JC      256   // GAT j-chunk staged in LDS

// CDNA5 async global->LDS path (ASYNCcnt-tracked), if this toolchain exposes it.
#if defined(__has_builtin)
#  if __has_builtin(__builtin_amdgcn_global_load_async_to_lds_b128) && \
      __has_builtin(__builtin_amdgcn_s_wait_asynccnt)
#    define USE_ASYNC_LDS 1
#  endif
#endif
#ifndef USE_ASYNC_LDS
#  define USE_ASYNC_LDS 0
#endif

#define GPTR(p) ((__attribute__((address_space(1))) b128_t*)(p))
#define LPTR(p) ((__attribute__((address_space(3))) b128_t*)(p))

// ---------------------------------------------------------------------------
// weight prep: f32 -> f16 (same layout)
__global__ void k_cvt(const float* __restrict__ in, _Float16* __restrict__ out, int n) {
  int i = blockIdx.x * 256 + threadIdx.x;
  if (i < n) out[i] = (_Float16)in[i];
}
// f32 [R][C] -> f16 [C][R]
__global__ void k_cvt_t(const float* __restrict__ in, _Float16* __restrict__ out, int R, int C) {
  int i = blockIdx.x * 256 + threadIdx.x;
  if (i >= R * C) return;
  int r = i / C, c = i % C;
  out[(size_t)c * R + r] = (_Float16)in[i];
}
// f32 [N][128] -> f16 [128][N] (activation transpose for GAT B-fragments)
__global__ void k_tr128(const float* __restrict__ in, _Float16* __restrict__ out, int N) {
  int i = blockIdx.x * 256 + threadIdx.x;
  if (i >= 128 * N) return;
  int d = i / N, n = i % N;
  out[i] = (_Float16)in[(size_t)n * 128 + d];
}

// ---------------------------------------------------------------------------
// input projection + positional encoding -> xh f16 [S*N][128], xlast f32 [N][128]
__global__ void k_inproj(const float* __restrict__ in, const float* __restrict__ w,
                         const float* __restrict__ b, _Float16* __restrict__ xh,
                         float* __restrict__ xlast, int N, int S) {
  int idx = blockIdx.x * 256 + threadIdx.x;
  if (idx >= S * N * DIM) return;
  int d = idx & 127;
  int t = idx >> 7;
  int s = t / N, n = t % N;
  const float* irow = in + ((size_t)n * S + s) * FIN_;
  const float* wrow = w + d * FIN_;
  float v = b[d];
#pragma unroll
  for (int f = 0; f < FIN_; ++f) v += irow[f] * wrow[f];
  int d2 = d & ~1;
  float div = __expf((float)d2 * (-9.210340371976184f / 128.0f));  // ln(10000)
  float ang = (float)s * div;
  v += (d & 1) ? __cosf(ang) : __sinf(ang);
  xh[idx] = (_Float16)v;
  if (s == S - 1) xlast[(size_t)n * 128 + d] = v;
}

// ---------------------------------------------------------------------------
// Generic WMMA GEMM:  C[M][Nout] = A[M][K] * W^T  (+bias, +relu)
// A f16 row-major; W f16 [Nout][K] row-major (so B = W^T, column-contiguous).
// One wave per 16x16 tile; 8 waves per block along M. M%128==0, Nout%16==0, K%32==0.
__global__ void k_gemm_f16(const _Float16* __restrict__ A, const _Float16* __restrict__ W,
                           const float* __restrict__ bias, float* __restrict__ Cf,
                           _Float16* __restrict__ Ch, int M, int K, int Nout, int relu) {
  int lane = threadIdx.x & 31;
  int wave = threadIdx.x >> 5;
  int m0 = (blockIdx.x * 8 + wave) * 16;
  int n0 = blockIdx.y * 16;
  int r  = lane & 15;   // A row within tile / B column within tile
  int hl = lane >> 4;   // lane half selects K sub-range per ISA fragment layout
  const _Float16* Arow = A + (size_t)(m0 + r) * K + hl * 8;
  const _Float16* Wcol = W + (size_t)(n0 + r) * K + hl * 8;
  v8f acc = {};
  for (int kb = 0; kb < K; kb += 32) {
    v8h alo = *(const v8h*)(Arow + kb);        // K = kb + hl*8 + [0..7]
    v8h ahi = *(const v8h*)(Arow + kb + 16);   // K = kb + 16 + hl*8 + [0..7]
    v8h blo = *(const v8h*)(Wcol + kb);        // K = kb + hl*8 + [0..7]
    v8h bhi = *(const v8h*)(Wcol + kb + 16);   // K = kb + 16 + hl*8 + [0..7]
    v16h a, b;
#pragma unroll
    for (int i = 0; i < 8; ++i) { a[i] = alo[i]; a[8 + i] = ahi[i]; b[i] = blo[i]; b[8 + i] = bhi[i]; }
    acc = __builtin_amdgcn_wmma_f32_16x16x32_f16(false, a, false, b, (short)0, acc, false, false);
  }
  float bv = bias ? bias[n0 + r] : 0.f;
#pragma unroll
  for (int v = 0; v < 8; ++v) {
    int row = m0 + v + 8 * hl;                 // C layout: M = vgpr + 8*(lane/16), N = lane%16
    float f = acc[v] + bv;
    if (relu) f = fmaxf(f, 0.f);
    size_t idx = (size_t)row * Nout + n0 + r;
    if (Cf) Cf[idx] = f;
    if (Ch) Ch[idx] = (_Float16)f;
  }
}

// ---------------------------------------------------------------------------
// transformer attention, last query position only. wave = (node, head), lane = t then d.
__global__ void k_attn(const float* __restrict__ q, const _Float16* __restrict__ kv,
                       _Float16* __restrict__ aoh, int N) {
  int n = blockIdx.x;
  int h = threadIdx.x >> 5;
  int lane = threadIdx.x & 31;
  __shared__ float qs[128];
  qs[threadIdx.x] = q[(size_t)n * 128 + threadIdx.x];
  __syncthreads();
  const _Float16* krow = kv + ((size_t)lane * N + n) * 256 + h * 32;  // lane = t
  float sc = 0.f;
#pragma unroll
  for (int j = 0; j < 32; ++j) sc += qs[h * 32 + j] * (float)krow[j];
  sc *= 0.17677669529663687f;  // 1/sqrt(32)
  float m = sc;
  for (int o = 16; o; o >>= 1) m = fmaxf(m, __shfl_xor(m, o, 32));
  float e = __expf(sc - m);
  float s = e;
  for (int o = 16; o; o >>= 1) s += __shfl_xor(s, o, 32);
  float a = e / s;
  float av = 0.f;                               // lane = d now
  for (int t = 0; t < 32; ++t) {
    float at = __shfl(a, t, 32);
    av += at * (float)kv[((size_t)t * N + n) * 256 + 128 + h * 32 + lane];
  }
  aoh[(size_t)n * 128 + h * 32 + lane] = (_Float16)av;
}

// ---------------------------------------------------------------------------
// y = LayerNorm(a + b) * g + be   (row = block, 128 threads)
__global__ void k_add_ln(const float* __restrict__ a, const float* __restrict__ b,
                         const float* __restrict__ g, const float* __restrict__ be,
                         float* __restrict__ yf, _Float16* __restrict__ yh) {
  int n = blockIdx.x, d = threadIdx.x;
  __shared__ float red[128];
  float v = a[(size_t)n * 128 + d] + b[(size_t)n * 128 + d];
  red[d] = v; __syncthreads();
  for (int o = 64; o; o >>= 1) { if (d < o) red[d] += red[d + o]; __syncthreads(); }
  float mu = red[0] * (1.f / 128.f); __syncthreads();
  float c = v - mu;
  red[d] = c * c; __syncthreads();
  for (int o = 64; o; o >>= 1) { if (d < o) red[d] += red[d + o]; __syncthreads(); }
  float y = c * rsqrtf(red[0] * (1.f / 128.f) + 1e-5f) * g[d] + be[d];
  yf[(size_t)n * 128 + d] = y;
  yh[(size_t)n * 128 + d] = (_Float16)y;
}

// ---------------------------------------------------------------------------
// GAT per-head attention coefficients  f1[h][n] = s_h[n]·wu_h,  f2[h][n] = s_h[n]·wv_h
__global__ void k_f1f2(const float* __restrict__ s, const float* __restrict__ wu,
                       const float* __restrict__ wv, float* __restrict__ f1,
                       float* __restrict__ f2, int N) {
  int idx = blockIdx.x * 256 + threadIdx.x;
  if (idx >= GH * N) return;
  int h = idx / N, n = idx % N;
  const float* srow = s + (size_t)n * 128 + h * GFO;
  float a = 0.f, b = 0.f;
#pragma unroll
  for (int f = 0; f < GFO; ++f) { a += srow[f] * wu[h * GFO + f]; b += srow[f] * wv[h * GFO + f]; }
  f1[h * N + n] = a;
  f2[h * N + n] = b;
}

// ---------------------------------------------------------------------------
// Fused GAT masked-softmax aggregation via WMMA.
// Block = 16-row i-tile, 8 waves = 8 heads. attn weights built in-register in
// the 16x32 f16 A-fragment layout; B = s^T fragments are contiguous loads.
// Tile staging into LDS uses the CDNA5 async global->LDS path when available.
__global__ void k_gat_agg(const float* __restrict__ adj, const float* __restrict__ f1,
                          const float* __restrict__ f2, const _Float16* __restrict__ sT,
                          float* __restrict__ out, int N) {
  __shared__ __attribute__((aligned(16))) float adjS[16][JC];
  __shared__ __attribute__((aligned(16))) float f1S[GH][JC];
  __shared__ float denomS[GH][16];
  int tid = threadIdx.x;
  int h = tid >> 5, lane = tid & 31;
  int r = lane & 15, hl = lane >> 4;
  int i0 = blockIdx.x * 16;
  float f2r = f2[h * N + i0 + r];
  v8f acc = {};
  float dsum = 0.f;
  for (int cb = 0; cb < N; cb += JC) {
    __syncthreads();
#if USE_ASYNC_LDS
    // 16 x JC f32 adjacency tile + GH x JC f1 tile, DMA'd straight to LDS
    for (int t = tid; t < 16 * (JC / 4); t += 256) {
      int rr = t / (JC / 4), jj = (t % (JC / 4)) * 4;
      __builtin_amdgcn_global_load_async_to_lds_b128(
          GPTR(&adj[(size_t)(i0 + rr) * N + cb + jj]), LPTR(&adjS[rr][jj]), 0, 0);
    }
    for (int t = tid; t < GH * (JC / 4); t += 256) {
      int hh = t / (JC / 4), jj = (t % (JC / 4)) * 4;
      __builtin_amdgcn_global_load_async_to_lds_b128(
          GPTR(&f1[(size_t)hh * N + cb + jj]), LPTR(&f1S[hh][jj]), 0, 0);
    }
    __builtin_amdgcn_s_wait_asynccnt(0);
#else
    for (int t = tid; t < 16 * (JC / 4); t += 256) {
      int rr = t / (JC / 4), jj = (t % (JC / 4)) * 4;
      *(float4*)&adjS[rr][jj] = *(const float4*)&adj[(size_t)(i0 + rr) * N + cb + jj];
    }
    for (int t = tid; t < GH * (JC / 4); t += 256) {
      int hh = t / (JC / 4), jj = (t % (JC / 4)) * 4;
      *(float4*)&f1S[hh][jj] = *(const float4*)&f1[(size_t)hh * N + cb + jj];
    }
#endif
    // prefetch next chunk's adjacency rows into cache while we compute
    int nb = cb + JC;
    if (nb < N && tid < 128)
      __builtin_prefetch(&adj[(size_t)(i0 + (tid >> 3)) * N + nb + ((tid & 7) << 5)], 0, 0);
    __syncthreads();
    for (int jb = 0; jb < JC; jb += 32) {
      v16h a;
#pragma unroll
      for (int i = 0; i < 16; ++i) {
        int k = hl * 16 + i;  // contiguous K map, matches B load
        int j = jb + k;
        float l = f1S[h][j] + f2r;
        l = (l < 0.f) ? 0.2f * l : l;                               // leaky_relu(0.2)
        float wv = (adjS[r][j] != 0.f) ? __expf(l) : 0.f;           // masked exp
        a[i] = (_Float16)wv;
        dsum += wv;
      }
      v16h b = *(const v16h*)(sT + (size_t)(h * GFO + r) * N + cb + jb + hl * 16);
      acc = __builtin_amdgcn_wmma_f32_16x16x32_f16(false, a, false, b, (short)0, acc, false, false);
    }
  }
  dsum += __shfl_xor(dsum, 16, 32);   // lanes r and r+16 jointly cover each row's K
  if (hl == 0) denomS[h][r] = dsum;
  __syncthreads();
#pragma unroll
  for (int v = 0; v < 8; ++v) {
    int row = v + 8 * hl;
    float den = denomS[h][row];
    float val = (den > 0.f) ? acc[v] / den : 0.f;
    out[(size_t)(i0 + row) * 128 + h * GFO + r] = val;
  }
}

// ---------------------------------------------------------------------------
// gat_out + bias(broadcast) + support-projection  -> f16
__global__ void k_combine(const float* __restrict__ g, const float* __restrict__ bvec,
                          const float* __restrict__ sp, _Float16* __restrict__ oh, int total) {
  int i = blockIdx.x * 256 + threadIdx.x;
  if (i >= total) return;
  oh[i] = (_Float16)(g[i] + bvec[i & 127] + sp[i]);
}

// ---------------------------------------------------------------------------
// semantic attention over 3 relation embeddings -> e[N][128]
__global__ void k_sem(const float* __restrict__ s0, const float* __restrict__ s1,
                      const float* __restrict__ s2, const float* __restrict__ w1,
                      const float* __restrict__ b1, const float* __restrict__ w2,
                      float* __restrict__ e) {
  int n = blockIdx.x, d = threadIdx.x;
  __shared__ float emb[3][128];
  __shared__ float red[128];
  emb[0][d] = s0[(size_t)n * 128 + d];
  emb[1][d] = s1[(size_t)n * 128 + d];
  emb[2][d] = s2[(size_t)n * 128 + d];
  __syncthreads();
  float wr[3];
  for (int rr = 0; rr < 3; ++rr) {
    float acc = b1[d];
    const float* row = w1 + (size_t)d * 128;
    for (int k = 0; k < 128; ++k) acc += emb[rr][k] * row[k];
    red[d] = tanhf(acc) * w2[d]; __syncthreads();
    for (int o = 64; o; o >>= 1) { if (d < o) red[d] += red[d + o]; __syncthreads(); }
    wr[rr] = red[0]; __syncthreads();
  }
  float mx = fmaxf(wr[0], fmaxf(wr[1], wr[2]));
  float e0 = __expf(wr[0] - mx), e1 = __expf(wr[1] - mx), e2 = __expf(wr[2] - mx);
  float inv = 1.f / (e0 + e1 + e2);
  e[(size_t)n * 128 + d] = (e0 * emb[0][d] + e1 * emb[1][d] + e2 * emb[2][d]) * inv;
}

__global__ void k_colmean(const float* __restrict__ e, float* __restrict__ mean, int N) {
  int d = blockIdx.x, tid = threadIdx.x;
  __shared__ float red[256];
  float s = 0.f;
  for (int n = tid; n < N; n += 256) s += e[(size_t)n * 128 + d];
  red[tid] = s; __syncthreads();
  for (int o = 128; o; o >>= 1) { if (tid < o) red[tid] += red[tid + o]; __syncthreads(); }
  if (!tid) mean[d] = red[0] / (float)N;
}

// PairNorm PN-SI + predictor + tanh
__global__ void k_final(const float* __restrict__ e, const float* __restrict__ mean,
                        const float* __restrict__ pw, const float* __restrict__ pb,
                        float* __restrict__ out) {
  int n = blockIdx.x, d = threadIdx.x;
  __shared__ float red[128];
  float v = e[(size_t)n * 128 + d] - mean[d];
  red[d] = v * v; __syncthreads();
  for (int o = 64; o; o >>= 1) { if (d < o) red[d] += red[d + o]; __syncthreads(); }
  float inv = rsqrtf(1e-6f + red[0]); __syncthreads();
  red[d] = v * inv * pw[d]; __syncthreads();
  for (int o = 64; o; o >>= 1) { if (d < o) red[d] += red[d + o]; __syncthreads(); }
  if (!d) out[n] = tanhf(red[0] + pb[0]);
}

// ---------------------------------------------------------------------------
extern "C" void kernel_launch(void* const* d_in, const int* in_sizes, int n_in,
                              void* d_out, int out_size, void* d_ws, size_t ws_size,
                              hipStream_t stream) {
  (void)in_sizes; (void)n_in; (void)out_size; (void)ws_size;
  const int N = N_NODES, S = SEQ, D = DIM;
  const size_t NB = (size_t)N * D;

  const float* inputs  = (const float*)d_in[0];
  const float* pos_adj = (const float*)d_in[1];
  const float* neg_adj = (const float*)d_in[2];
  const float* ip_w  = (const float*)d_in[3];
  const float* ip_b  = (const float*)d_in[4];
  const float* in_w  = (const float*)d_in[5];
  const float* in_b  = (const float*)d_in[6];
  const float* out_w = (const float*)d_in[7];
  const float* out_b = (const float*)d_in[8];
  const float* l1_w  = (const float*)d_in[9];
  const float* l1_b  = (const float*)d_in[10];
  const float* l2_w  = (const float*)d_in[11];
  const float* l2_b  = (const float*)d_in[12];
  const float* n1_g  = (const float*)d_in[13];
  const float* n1_b  = (const float*)d_in[14];
  const float* n2_g  = (const float*)d_in[15];
  const float* n2_b  = (const float*)d_in[16];
  const float* self_w = (const float*)d_in[17];
  const float* self_b = (const float*)d_in[18];
  const float* pmlp_w = (const float*)d_in[19];
  const float* pmlp_b = (const float*)d_in[20];
  const float* nmlp_w = (const float*)d_in[21];
  const float* nmlp_b = (const float*)d_in[22];
  const float* sem_w1 = (const float*)d_in[23];
  const float* sem_b1 = (const float*)d_in[24];
  const float* sem_w2 = (const float*)d_in[25];
  const float* pred_w = (const float*)d_in[26];
  const float* pred_b = (const float*)d_in[27];
  const float* pos_W  = (const float*)d_in[28];
  const float* pos_wu = (const float*)d_in[29];
  const float* pos_wv = (const float*)d_in[30];
  const float* pos_bb = (const float*)d_in[31];
  const float* pproj_w = (const float*)d_in[32];
  const float* pproj_b = (const float*)d_in[33];
  const float* neg_W  = (const float*)d_in[34];
  const float* neg_wu = (const float*)d_in[35];
  const float* neg_wv = (const float*)d_in[36];
  const float* neg_bb = (const float*)d_in[37];
  const float* nproj_w = (const float*)d_in[38];
  const float* nproj_b = (const float*)d_in[39];
  float* out = (float*)d_out;

  // ---- workspace layout (bump alloc with region reuse) ----
  char* ws = (char*)d_ws;
  const size_t SZ_XH  = (size_t)S * N * D * 2;      // 25.2 MB
  const size_t SZ_KVH = (size_t)S * N * 2 * D * 2;  // 50.3 MB
  char* regA = ws;                 // xh; reused after QKV GEMMs
  char* regB = ws + SZ_XH;         // kvh; reused after attention
  char* regC = ws + SZ_XH + SZ_KVH;

  _Float16* xh  = (_Float16*)regA;
  _Float16* kvh = (_Float16*)regB;

  size_t oA = 0, oB = 0, oC = 0;
  auto A_ = [&](size_t b) { void* p = regA + oA; oA += (b + 255) & ~(size_t)255; return p; };
  auto B_ = [&](size_t b) { void* p = regB + oB; oB += (b + 255) & ~(size_t)255; return p; };
  auto C_ = [&](size_t b) { void* p = regC + oC; oC += (b + 255) & ~(size_t)255; return p; };

  // pool A (live only after xh is dead)
  _Float16* aoh   = (_Float16*)A_(NB * 2);
  float*    aopf  = (float*)A_(NB * 4);
  float*    x1f   = (float*)A_(NB * 4);
  _Float16* x1h   = (_Float16*)A_(NB * 2);
  _Float16* h1h   = (_Float16*)A_((size_t)N * DFF_ * 2);
  float*    h2f   = (float*)A_(NB * 4);
  float*    suppf = (float*)A_(NB * 4);
  _Float16* supph = (_Float16*)A_(NB * 2);
  float*    sposf = (float*)A_(NB * 4);
  float*    snegf = (float*)A_(NB * 4);
  _Float16* sTpos = (_Float16*)A_(NB * 2);
  _Float16* sTneg = (_Float16*)A_(NB * 2);
  float*    f1p   = (float*)A_((size_t)GH * N * 4);
  float*    f2p   = (float*)A_((size_t)GH * N * 4);
  float*    f1n   = (float*)A_((size_t)GH * N * 4);
  float*    f2n   = (float*)A_((size_t)GH * N * 4);
  // pool B (live only after kvh is dead)
  float*    gposf  = (float*)B_(NB * 4);
  float*    gnegf  = (float*)B_(NB * 4);
  float*    sprojp = (float*)B_(NB * 4);
  float*    sprojn = (float*)B_(NB * 4);
  _Float16* possh  = (_Float16*)B_(NB * 2);
  _Float16* negsh  = (_Float16*)B_(NB * 2);
  float*    s0f    = (float*)B_(NB * 4);
  float*    s1f    = (float*)B_(NB * 4);
  float*    s2f    = (float*)B_(NB * 4);
  float*    ef     = (float*)B_(NB * 4);
  float*    meanv  = (float*)B_(512);
  // region C (persistent)
  float*    xlast = (float*)C_(NB * 4);
  float*    qf    = (float*)C_(NB * 4);
  _Float16* inwh  = (_Float16*)C_((size_t)384 * 128 * 2);
  _Float16* outwh = (_Float16*)C_((size_t)D * D * 2);
  _Float16* l1h   = (_Float16*)C_((size_t)DFF_ * D * 2);
  _Float16* l2h   = (_Float16*)C_((size_t)D * DFF_ * 2);
  _Float16* selfh = (_Float16*)C_((size_t)D * D * 2);
  _Float16* pmlph = (_Float16*)C_((size_t)D * D * 2);
  _Float16* nmlph = (_Float16*)C_((size_t)D * D * 2);
  _Float16* pprojh = (_Float16*)C_((size_t)D * D * 2);
  _Float16* nprojh = (_Float16*)C_((size_t)D * D * 2);
  _Float16* posWt = (_Float16*)C_((size_t)D * D * 2);
  _Float16* negWt = (_Float16*)C_((size_t)D * D * 2);

  auto cvt = [&](const float* src, _Float16* dst, int n) {
    k_cvt<<<(n + 255) / 256, 256, 0, stream>>>(src, dst, n);
  };
  auto gemm = [&](const _Float16* A, const _Float16* W, const float* bias,
                  float* Cf, _Float16* Ch, int M, int K, int Nout, int relu) {
    dim3 g(M / 128, Nout / 16);
    k_gemm_f16<<<g, 256, 0, stream>>>(A, W, bias, Cf, Ch, M, K, Nout, relu);
  };

  // 1) weight prep
  cvt(in_w,  inwh, 384 * 128);
  cvt(out_w, outwh, D * D);
  cvt(l1_w,  l1h, DFF_ * D);
  cvt(l2_w,  l2h, D * DFF_);
  cvt(self_w, selfh, D * D);
  cvt(pmlp_w, pmlph, D * D);
  cvt(nmlp_w, nmlph, D * D);
  cvt(pproj_w, pprojh, D * D);
  cvt(nproj_w, nprojh, D * D);
  k_cvt_t<<<(D * D + 255) / 256, 256, 0, stream>>>(pos_W, posWt, D, D);
  k_cvt_t<<<(D * D + 255) / 256, 256, 0, stream>>>(neg_W, negWt, D, D);

  // 2) input projection + PE
  k_inproj<<<(S * N * D + 255) / 256, 256, 0, stream>>>(inputs, ip_w, ip_b, xh, xlast, N, S);

  // 3) KV for all tokens, Q for last token only (dead-code eliminated transformer)
  gemm(xh, inwh + 128 * 128, in_b + 128, nullptr, kvh, S * N, D, 2 * D, 0);
  gemm(xh + (size_t)(S - 1) * N * D, inwh, in_b, qf, nullptr, N, D, D, 0);

  // 4) attention (last position), 5) out-proj, 6) LN1
  k_attn<<<N, 128, 0, stream>>>(qf, kvh, aoh, N);
  gemm(aoh, outwh, out_b, aopf, nullptr, N, D, D, 0);
  k_add_ln<<<N, 128, 0, stream>>>(xlast, aopf, n1_g, n1_b, x1f, x1h);

  // 7) FFN + LN2 -> support
  gemm(x1h, l1h, l1_b, nullptr, h1h, N, D, DFF_, 1);
  gemm(h1h, l2h, l2_b, h2f, nullptr, N, DFF_, D, 0);
  k_add_ln<<<N, 128, 0, stream>>>(x1f, h2f, n2_g, n2_b, suppf, supph);

  // 8) GAT features s = support @ W (per relation), transposed f16 copies
  gemm(supph, posWt, nullptr, sposf, nullptr, N, D, D, 0);
  gemm(supph, negWt, nullptr, snegf, nullptr, N, D, D, 0);
  k_tr128<<<(128 * N + 255) / 256, 256, 0, stream>>>(sposf, sTpos, N);
  k_tr128<<<(128 * N + 255) / 256, 256, 0, stream>>>(snegf, sTneg, N);

  // 9) attention coefficients + fused masked-softmax aggregation (WMMA)
  k_f1f2<<<(GH * N + 255) / 256, 256, 0, stream>>>(sposf, pos_wu, pos_wv, f1p, f2p, N);
  k_f1f2<<<(GH * N + 255) / 256, 256, 0, stream>>>(snegf, neg_wu, neg_wv, f1n, f2n, N);
  k_gat_agg<<<N / 16, 256, 0, stream>>>(pos_adj, f1p, f2p, sTpos, gposf, N);
  k_gat_agg<<<N / 16, 256, 0, stream>>>(neg_adj, f1n, f2n, sTneg, gnegf, N);

  // 10) residual projections + combine -> pos_s / neg_s (f16)
  gemm(supph, pprojh, pproj_b, sprojp, nullptr, N, D, D, 0);
  gemm(supph, nprojh, nproj_b, sprojn, nullptr, N, D, D, 0);
  k_combine<<<((int)NB + 255) / 256, 256, 0, stream>>>(gposf, pos_bb, sprojp, possh, (int)NB);
  k_combine<<<((int)NB + 255) / 256, 256, 0, stream>>>(gnegf, neg_bb, sprojn, negsh, (int)NB);

  // 11) relation MLPs
  gemm(supph, selfh, self_b, s0f, nullptr, N, D, D, 0);
  gemm(possh, pmlph, pmlp_b, s1f, nullptr, N, D, D, 0);
  gemm(negsh, nmlph, nmlp_b, s2f, nullptr, N, D, D, 0);

  // 12) semantic attention, PairNorm, predictor
  k_sem<<<N, 128, 0, stream>>>(s0f, s1f, s2f, sem_w1, sem_b1, sem_w2, ef);
  k_colmean<<<128, 256, 0, stream>>>(ef, meanv, N);
  k_final<<<N, 128, 0, stream>>>(ef, meanv, pred_w, pred_b, out);
}
